// CAMIL_26920855011670
// MI455X (gfx1250) — compile-verified, matrix operand order checked by reading, not run on hardware
//
#include <hip/hip_runtime.h>
#include <hip/hip_bf16.h>

#define NN    10000
#define DD    512
#define NPAD  10240
#define PAD   240
#define HEADS 8
#define DH    64
#define LM    256

typedef __bf16 bf16_t;
typedef __attribute__((ext_vector_type(16))) __bf16 v16bf;
typedef __attribute__((ext_vector_type(8)))  float  v8f;
typedef __attribute__((ext_vector_type(2)))  float  v2f;

// ---------------- WMMA fragment helper (wave32, CDNA5 layout) ----------------
// A (or B^T) fragment from row-major storage, row = base+(lane&15), K pattern:
//   lanes 0-15:  K = {kb+0..7, kb+16..23};  lanes 16-31: K = {kb+8..15, kb+24..31}
// Two contiguous 8-float runs -> two float4 pairs (b128 loads), cvt to bf16.
__device__ __forceinline__ v16bf frag_rm(const float* src, int ld, int row, int kb, int kh) {
  const float4* p4 = (const float4*)(src + (long)row * ld + kb + kh * 8);
  float4 x0 = p4[0], x1 = p4[1], x2 = p4[4], x3 = p4[5];
  v16bf f;
  f[0]  = (bf16_t)x0.x; f[1]  = (bf16_t)x0.y; f[2]  = (bf16_t)x0.z; f[3]  = (bf16_t)x0.w;
  f[4]  = (bf16_t)x1.x; f[5]  = (bf16_t)x1.y; f[6]  = (bf16_t)x1.z; f[7]  = (bf16_t)x1.w;
  f[8]  = (bf16_t)x2.x; f[9]  = (bf16_t)x2.y; f[10] = (bf16_t)x2.z; f[11] = (bf16_t)x2.w;
  f[12] = (bf16_t)x3.x; f[13] = (bf16_t)x3.y; f[14] = (bf16_t)x3.z; f[15] = (bf16_t)x3.w;
  return f;
}
__device__ __forceinline__ v8f wmma_bf16(v16bf a, v16bf b, v8f c) {
  return __builtin_amdgcn_wmma_f32_16x16x32_bf16(false, a, false, b, (short)0, c, false, false);
}

// ---------------- Generic batched bf16 WMMA GEMM (B supplied as N x K) ----------------
// Block = 8 waves arranged 4(M) x 2(N); each wave owns a 32x32 micro-tile
// (2 A-frags x 2 B-frags -> 4 wmma per K-step: 2x register-level reuse).
struct GemmP {
  const float* A; long sAb; int lda; int arow0;        // A logical row m -> physical m+arow0 (<0 => zero tile)
  const float* B; long sBb; int ldb;                   // B^T: N x K row-major
  float*       C; long sCb; int ldc; int ct;           // ct: store transposed (C[n*ldc+m])
  const float* bias;
  const float* E; long sEb; int ldE; float beta;
  float alpha;
  int M, N, Kd;
  int mode;                                            // 0 = plain, 1 = QKV scatter
  float* Qo; float* Ko; float* Vo;
};

__global__ void gemm_bf16_kernel(GemmP p) {
  int lane = threadIdx.x & 31, wave = threadIdx.x >> 5;
  int lm = lane & 15, kh = lane >> 4;
  int wm = wave >> 1, wn = wave & 1;
  int m0 = blockIdx.x * 128 + wm * 32;
  if (m0 >= p.M) return;
  int n0 = blockIdx.y * 64 + wn * 32;
  int b  = blockIdx.z;
  const float* A = p.A + (long)b * p.sAb;
  const float* B = p.B + (long)b * p.sBb;
  // sub-tile guards (wave-uniform; PAD and M are multiples of 16)
  int okc0 = __builtin_amdgcn_readfirstlane((m0 + p.arow0 >= 0) ? 1 : 0);
  int st1  = __builtin_amdgcn_readfirstlane((m0 + 16 < p.M) ? 1 : 0);
  int okc1 = __builtin_amdgcn_readfirstlane(((m0 + 16 + p.arow0 >= 0) && (m0 + 16 < p.M)) ? 1 : 0);
  v8f acc00 = {}, acc01 = {}, acc10 = {}, acc11 = {};
  int ar0 = m0 + lm + p.arow0, ar1 = ar0 + 16;
  int nr0 = n0 + lm,            nr1 = nr0 + 16;
  for (int kb = 0; kb < p.Kd; kb += 32) {
    v16bf b0 = frag_rm(B, p.ldb, nr0, kb, kh);
    v16bf b1 = frag_rm(B, p.ldb, nr1, kb, kh);
    if (okc0) {
      v16bf a0 = frag_rm(A, p.lda, ar0, kb, kh);
      acc00 = wmma_bf16(a0, b0, acc00);
      acc01 = wmma_bf16(a0, b1, acc01);
    }
    if (okc1) {
      v16bf a1 = frag_rm(A, p.lda, ar1, kb, kh);
      acc10 = wmma_bf16(a1, b0, acc10);
      acc11 = wmma_bf16(a1, b1, acc11);
    }
  }
  v8f accs[2][2] = {{acc00, acc01}, {acc10, acc11}};
  if (p.mode == 0) {
#pragma unroll
    for (int i = 0; i < 2; ++i) {
      if (i == 1 && !st1) continue;
#pragma unroll
      for (int j = 0; j < 2; ++j) {
        int nc = n0 + j * 16 + lm;
        float bv = p.bias ? p.bias[nc] : 0.f;
#pragma unroll
        for (int r = 0; r < 8; ++r) {
          int m = m0 + i * 16 + r + 8 * kh;
          float v = p.alpha * accs[i][j][r] + bv;
          if (p.E) v += p.beta * p.E[(long)b * p.sEb + (long)m * p.ldE + nc];
          if (!p.ct) p.C[(long)b * p.sCb + (long)m * p.ldc + nc] = v;
          else       p.C[(long)b * p.sCb + (long)nc * p.ldc + m] = v;
        }
      }
    }
  } else {                                             // QKV scatter: col j -> (which, head, dh)
#pragma unroll
    for (int j = 0; j < 2; ++j) {
      int jc = n0 + j * 16 + lm;
      int which = jc >> 9, hd = jc & 511, head = hd >> 6, dh = hd & 63;
      float scale = (which == 0) ? 0.125f : 1.0f;      // q * dim_head^-0.5
      float* dst = (which == 0 ? p.Qo : which == 1 ? p.Ko : p.Vo) + (long)head * NPAD * DH + dh;
#pragma unroll
      for (int i = 0; i < 2; ++i) {
        if (i == 1 && !st1) continue;
#pragma unroll
        for (int r = 0; r < 8; ++r) {
          int m = m0 + i * 16 + r + 8 * kh;
          dst[(long)m * DH] = scale * accs[i][j][r];
        }
      }
    }
  }
}

// ---------------- 256x256x256 f32 WMMA matmul, D = alpha*(A@B) + beta*E ----------------
// B panel staged transposed through LDS; A rows read as float4.
__global__ void mm256_kernel(const float* A, const float* B, const float* E, float* D,
                             float alpha, float beta) {
  __shared__ __align__(16) float smB[16 * 272];
  int lane = threadIdx.x & 31, wave = threadIdx.x >> 5;
  int lm = lane & 15, kh = lane >> 4;
  int m0 = blockIdx.x * 128 + wave * 16;
  int n0 = blockIdx.y * 16;
  long off = (long)blockIdx.z * (LM * LM);
  const float* Ab = A + off; const float* Bb = B + off;
  for (int i = threadIdx.x; i < 16 * LM; i += 256) {   // stage B(256x16) -> smB[n][k]
    int k = i >> 4, c = i & 15;
    smB[c * 272 + k] = Bb[(long)k * LM + n0 + c];
  }
  __syncthreads();
  v8f acc = {};
#if __has_builtin(__builtin_amdgcn_wmma_f32_16x16x4_f32)
  const float4* arow = (const float4*)(Ab + (long)(m0 + lm) * LM);
  const float* brow = smB + lm * 272 + 2 * kh;
#pragma unroll 8
  for (int j = 0; j < 64; ++j) {
    float4 av = arow[j];
    v2f a, b;
    a[0] = kh ? av.z : av.x;
    a[1] = kh ? av.w : av.y;
    b[0] = brow[4 * j];
    b[1] = brow[4 * j + 1];
    acc = __builtin_amdgcn_wmma_f32_16x16x4_f32(false, a, false, b, (short)0, acc, false, false);
  }
#else
  for (int kb = 0; kb < LM; kb += 32) {
    v16bf a = frag_rm(Ab, LM, m0 + lm, kb, kh);
    v16bf b = frag_rm(smB, 272, lm, kb, kh);
    acc = wmma_bf16(a, b, acc);
  }
#endif
#pragma unroll
  for (int r = 0; r < 8; ++r) {
    int m = m0 + r + 8 * kh;
    float v = alpha * acc[r];
    if (E) v += beta * E[off + (long)m * LM + n0 + lm];
    D[off + (long)m * LM + n0 + lm] = v;
  }
}

// ---------------- tiled transpose: out(cols x rows) = in(rows x cols)^T, batched ----------------
__global__ void transpose_kernel(const float* in, float* out, int rows, int cols) {
  __shared__ float tile[32][33];
  long ib = (long)blockIdx.z * rows * cols;
  int r0 = blockIdx.y * 32, c0 = blockIdx.x * 32;
  int tx = threadIdx.x & 31, ty = threadIdx.x >> 5;
  for (int dy = ty; dy < 32; dy += 8)
    tile[dy][tx] = in[ib + (long)(r0 + dy) * cols + (c0 + tx)];
  __syncthreads();
  for (int dy = ty; dy < 32; dy += 8)
    out[ib + (long)(c0 + dy) * rows + (r0 + tx)] = tile[tx][dy];
}

// ---------------- attn2 = softmax_rows(q_l @ k_l^T), per head ----------------
__global__ void attn2_kernel(const float* QL, const float* KL, float* X2) {
  __shared__ __align__(16) float sm[16 * 256];
  __shared__ float rinv[16];
  int head = blockIdx.y, m0 = blockIdx.x * 16;
  const float* Aq = QL + (long)head * LM * DH;
  const float* Bk = KL + (long)head * LM * DH;
  int lane = threadIdx.x & 31, wave = threadIdx.x >> 5;
  int lm = lane & 15, kh = lane >> 4;
  for (int t = wave; t < 16; t += 8) {
    int n0 = t * 16;
    v8f acc = {};
#pragma unroll
    for (int kb = 0; kb < DH; kb += 32) {
      v16bf a = frag_rm(Aq, DH, m0 + lm, kb, kh);
      v16bf b = frag_rm(Bk, DH, n0 + lm, kb, kh);
      acc = wmma_bf16(a, b, acc);
    }
#pragma unroll
    for (int r = 0; r < 8; ++r) sm[(r + 8 * kh) * 256 + n0 + lm] = acc[r];
  }
  __syncthreads();
  if (threadIdx.x < 16) {
    int r = threadIdx.x;
    float mx = -1e30f;
    for (int j = 0; j < 256; ++j) mx = fmaxf(mx, sm[r * 256 + j]);
    float s = 0.f;
    for (int j = 0; j < 256; ++j) { float e = __expf(sm[r * 256 + j] - mx); sm[r * 256 + j] = e; s += e; }
    rinv[r] = 1.f / s;
  }
  __syncthreads();
  for (int idx = threadIdx.x; idx < 16 * 256; idx += 256) {
    int r = idx >> 8;
    X2[(long)head * LM * LM + (long)(m0 + r) * LM + (idx & 255)] = sm[idx] * rinv[r];
  }
}

// ---------------- fused: out16x64 = softmax_rows(Q tile @ k_l^T) @ T4, scatter to concat ----------------
// T4 supplied transposed (64 x 256 per head).
__global__ void attn1_fused_kernel(const float* Q, const float* KL, const float* T4T, float* OC) {
  __shared__ __align__(16) float sm[16 * 256];
  __shared__ float rinv[16];
  int head = blockIdx.y, m0 = blockIdx.x * 16;
  const float* Aq   = Q   + (long)head * NPAD * DH + (long)PAD * DH;  // only last NN rows survive
  const float* Bk   = KL  + (long)head * LM * DH;
  const float* T4Th = T4T + (long)head * DH * LM;
  int lane = threadIdx.x & 31, wave = threadIdx.x >> 5;
  int lm = lane & 15, kh = lane >> 4;
  for (int t = wave; t < 16; t += 8) {
    int n0 = t * 16;
    v8f acc = {};
#pragma unroll
    for (int kb = 0; kb < DH; kb += 32) {
      v16bf a = frag_rm(Aq, DH, m0 + lm, kb, kh);
      v16bf b = frag_rm(Bk, DH, n0 + lm, kb, kh);
      acc = wmma_bf16(a, b, acc);
    }
#pragma unroll
    for (int r = 0; r < 8; ++r) sm[(r + 8 * kh) * 256 + n0 + lm] = acc[r];
  }
  __syncthreads();
  if (threadIdx.x < 16) {
    int r = threadIdx.x;
    float mx = -1e30f;
    for (int j = 0; j < 256; ++j) mx = fmaxf(mx, sm[r * 256 + j]);
    float s = 0.f;
    for (int j = 0; j < 256; ++j) { float e = __expf(sm[r * 256 + j] - mx); sm[r * 256 + j] = e; s += e; }
    rinv[r] = 1.f / s;
  }
  __syncthreads();
  if (wave < 4) {                                      // P(16x256) @ T4(256x64)
    int n0 = wave * 16;
    v8f acc = {};
    for (int kb = 0; kb < 256; kb += 32) {
      v16bf a = frag_rm(sm, 256, lm, kb, kh);
      v16bf b = frag_rm(T4Th, LM, n0 + lm, kb, kh);
      acc = wmma_bf16(a, b, acc);
    }
#pragma unroll
    for (int r = 0; r < 8; ++r) {
      int rl = r + 8 * kh;
      OC[(long)(m0 + rl) * DD + head * DH + n0 + lm] = acc[r] * rinv[rl];
    }
  }
}

// ---------------- small helper kernels ----------------
__global__ void init_kernel(float* out, unsigned int* mb) {
  int i = blockIdx.x * 256 + threadIdx.x;
  if (i < NN) { out[1 + i] = 0.f; out[1 + NN + i] = 1.f; }  // A_raw=0, k_alpha=1 (softmax over size-1 axis)
  if (i == 0) { out[0] = 0.f; mb[0] = 0u; mb[1] = 0u; }
}

__global__ void landmark_kernel(const float* Q, const float* K, float* QL, float* KL) {
  int i = blockIdx.x * 256 + threadIdx.x;                   // HEADS*LM*DH = 131072
  if (i >= HEADS * LM * DH) return;
  int dh = i & 63, j = (i >> 6) & 255, h = i >> 14;
  const float* q = Q + (long)h * NPAD * DH + (long)j * 40 * DH + dh;
  const float* k = K + (long)h * NPAD * DH + (long)j * 40 * DH + dh;
  float sq = 0.f, sk = 0.f;
  for (int t = 0; t < 40; ++t) { sq += q[(long)t * DH]; sk += k[(long)t * DH]; }
  QL[i] = sq * (1.f / 40.f); KL[i] = sk * (1.f / 40.f);
}

__global__ void pinvscale_kernel(const float* X, unsigned int* mb) {
  __shared__ float r1[256], r2[256];
  int h = blockIdx.x >> 8, r = blockIdx.x & 255, t = threadIdx.x;
  const float* base = X + (long)h * LM * LM;
  r1[t] = fabsf(base[(long)r * LM + t]);
  r2[t] = fabsf(base[(long)t * LM + r]);
  __syncthreads();
  for (int k = 128; k; k >>= 1) { if (t < k) { r1[t] += r1[t + k]; r2[t] += r2[t + k]; } __syncthreads(); }
  if (t == 0) {
    atomicMax((int*)&mb[0], __float_as_int(r1[0]));   // max row-sum ("col" in ref)
    atomicMax((int*)&mb[1], __float_as_int(r2[0]));   // max col-sum ("row" in ref)
  }
}

__global__ void zinit_kernel(const float* X, const unsigned int* mb, float* Z) {
  long i = (long)blockIdx.x * 256 + threadIdx.x;
  if (i >= (long)HEADS * LM * LM) return;
  int h = (int)(i >> 16), rest = (int)(i & 65535), row = rest >> 8, col = rest & 255;
  float s = __int_as_float((int)mb[0]) * __int_as_float((int)mb[1]);
  Z[i] = X[(long)h * LM * LM + (long)col * LM + row] / s;
}

__global__ void rowsm_kernel(float* X, int cols) {
  __shared__ float red[256];
  __shared__ float smx, sinv;
  long base = (long)blockIdx.x * cols;
  int t = threadIdx.x;
  float mx = -1e30f;
  for (int j = t; j < cols; j += 256) mx = fmaxf(mx, X[base + j]);
  red[t] = mx; __syncthreads();
  for (int k = 128; k; k >>= 1) { if (t < k) red[t] = fmaxf(red[t], red[t + k]); __syncthreads(); }
  if (!t) smx = red[0];
  __syncthreads();
  mx = smx;
  float s = 0.f;
  for (int j = t; j < cols; j += 256) s += __expf(X[base + j] - mx);
  __syncthreads();
  red[t] = s; __syncthreads();
  for (int k = 128; k; k >>= 1) { if (t < k) red[t] += red[t + k]; __syncthreads(); }
  if (!t) sinv = 1.f / red[0];
  __syncthreads();
  float inv = sinv;
  for (int j = t; j < cols; j += 256) X[base + j] = __expf(X[base + j] - mx) * inv;
}

__global__ void convres_kernel(const float* V, const float* Wc, float* OC) {
  long i = (long)blockIdx.x * 256 + threadIdx.x;
  if (i >= (long)NN * DD) return;
  int n = (int)(i >> 9), c = (int)(i & 511);
  int head = c >> 6, dh = c & 63;
  const float* v = V + (long)head * NPAD * DH + dh;
  const float* wp = Wc + head * 33;
  int base = PAD + n - 16;
  float s = 0.f;
#pragma unroll
  for (int t = 0; t < 33; ++t) {
    int nn2 = base + t;
    if (nn2 >= 0 && nn2 < NPAD) s += wp[t] * v[(long)nn2 * DH];
  }
  OC[i] += s;
}

__global__ void edge_kernel(const int* row, const int* col, const float* val,
                            const float* q2, const float* k2, float* Araw, int E) {
  int wid = (int)(((long)blockIdx.x * blockDim.x + threadIdx.x) >> 5);
  int lane = threadIdx.x & 31;
  if (wid >= E) return;
  int r = row[wid], c = col[wid];
  const float* qp = q2 + (long)r * 256;
  const float* kp = k2 + (long)c * 256;
  float s = 0.f;
  for (int t = lane; t < 256; t += 32) s += qp[t] * kp[t];
  for (int off = 16; off; off >>= 1) s += __shfl_xor(s, off, 32);
  if (lane == 0) atomicAdd(&Araw[r], val[wid] * s * 0.0625f);   // / sqrt(256)
}

__global__ void alpha_kernel(const float* Araw, float* alpha) {
  __shared__ float red[1024];
  __shared__ float smx, sinv;
  int t = threadIdx.x;
  float mx = -1e30f;
  for (int i = t; i < NN; i += 1024) mx = fmaxf(mx, Araw[i]);
  red[t] = mx; __syncthreads();
  for (int k = 512; k; k >>= 1) { if (t < k) red[t] = fmaxf(red[t], red[t + k]); __syncthreads(); }
  if (!t) smx = red[0];
  __syncthreads();
  mx = smx;
  float s = 0.f;
  for (int i = t; i < NN; i += 1024) s += __expf(Araw[i] - mx);
  __syncthreads();
  red[t] = s; __syncthreads();
  for (int k = 512; k; k >>= 1) { if (t < k) red[t] += red[t + k]; __syncthreads(); }
  if (!t) sinv = 1.f / red[0];
  __syncthreads();
  float inv = sinv;
  for (int i = t; i < NN; i += 1024) alpha[i] = __expf(Araw[i] - mx) * inv;
}

__global__ void xo_kernel(const float* value, const float* enc, const float* alpha, float* xo) {
  long i = (long)blockIdx.x * 256 + threadIdx.x;
  if (i >= (long)NN * DD) return;
  int n = (int)(i >> 9);
  float xl = alpha[n] * value[i];
  float sg = 1.f / (1.f + __expf(xl));                 // sigmoid(-xl)
  float w2 = sg * sg;
  xo[i] = 2.f * xl * w2 + 2.f * enc[i] * (1.f - w2);
}

__global__ void colsum_kernel(const float* xo, float* pooled) {
  __shared__ float red[256];
  int c = blockIdx.x, t = threadIdx.x;
  float s = 0.f;
  for (int n = t; n < NN; n += 256) s += xo[(long)n * DD + c];
  red[t] = s; __syncthreads();
  for (int k = 128; k; k >>= 1) { if (t < k) red[t] += red[t + k]; __syncthreads(); }
  if (!t) pooled[c] = red[0];
}

__global__ void final_kernel(const float* pooled, const float* fc, const float* fcb, float* out) {
  __shared__ float red[512];
  int t = threadIdx.x;
  red[t] = pooled[t] * fc[t]; __syncthreads();
  for (int k = 256; k; k >>= 1) { if (t < k) red[t] += red[t + k]; __syncthreads(); }
  if (!t) out[0] = red[0] + fcb[0];
}

// ---------------- host side ----------------
extern "C" void kernel_launch(void* const* d_in, const int* in_sizes, int n_in,
                              void* d_out, int out_size, void* d_ws, size_t ws_size,
                              hipStream_t stream) {
  (void)n_in; (void)out_size; (void)ws_size;
  const float* bag   = (const float*)d_in[0];
  const int*   arow  = (const int*)d_in[1];
  const int*   acol  = (const int*)d_in[2];
  const float* aval  = (const float*)d_in[3];
  const float* w_qkv = (const float*)d_in[4];
  const float* w_out = (const float*)d_in[5];
  const float* b_out = (const float*)d_in[6];
  const float* convw = (const float*)d_in[7];
  const float* wq    = (const float*)d_in[8];
  const float* wk    = (const float*)d_in[9];
  const float* wv_w  = (const float*)d_in[10];
  const float* wv_b  = (const float*)d_in[11];
  // d_in[12..14] (v_att,u_att,w_att) are dead: k_alpha == softmax over size-1 axis == 1
  const float* fc_k  = (const float*)d_in[15];
  const float* fc_b  = (const float*)d_in[16];
  int E = in_sizes[1];
  float* out = (float*)d_out;

  float* w = (float*)d_ws;
  size_t o = 0;
  auto alloc = [&](size_t n) { float* p = w + o; o += (n + 63) & ~(size_t)63; return p; };
  float* Q    = alloc((size_t)HEADS * NPAD * DH);
  float* Kb   = alloc((size_t)HEADS * NPAD * DH);
  float* Vb   = alloc((size_t)HEADS * NPAD * DH);
  float* VbT  = alloc((size_t)HEADS * DH * NPAD);
  float* QL   = alloc((size_t)HEADS * LM * DH);
  float* KL   = alloc((size_t)HEADS * LM * DH);
  float* X2   = alloc((size_t)HEADS * LM * LM);
  float* Z0   = alloc((size_t)HEADS * LM * LM);
  float* Z1   = alloc((size_t)HEADS * LM * LM);
  float* A1   = alloc((size_t)HEADS * LM * LM);
  float* T1   = alloc((size_t)HEADS * LM * LM);
  float* T2   = alloc((size_t)HEADS * LM * LM);
  float* AT3  = alloc((size_t)HEADS * LM * NPAD);
  float* t3T  = alloc((size_t)HEADS * DH * LM);
  float* t4T  = alloc((size_t)HEADS * DH * LM);
  float* OC   = alloc((size_t)NN * DD);
  float* ENC  = alloc((size_t)NN * DD);
  float* Q2   = alloc((size_t)NN * 256);
  float* K2   = alloc((size_t)NN * 256);
  float* VAL  = alloc((size_t)NN * DD);
  float* XO   = alloc((size_t)NN * DD);
  float* ALP  = alloc(NN);
  float* POOL = alloc(DD);
  float* WQKVT = alloc((size_t)3 * DD * DD);   // 1536 x 512
  float* WOUTT = alloc((size_t)DD * DD);
  float* WQT   = alloc((size_t)256 * DD);
  float* WKT   = alloc((size_t)256 * DD);
  float* WVWT  = alloc((size_t)DD * DD);
  unsigned int* MB = (unsigned int*)alloc(16);

  init_kernel<<<40, 256, 0, stream>>>(out, MB);

  // 0) one-pass weight transposes so all GEMM B-operands are N x K (vectorized loads)
  transpose_kernel<<<dim3(48, 16, 1), 256, 0, stream>>>(w_qkv, WQKVT, DD, 3 * DD);
  transpose_kernel<<<dim3(16, 16, 1), 256, 0, stream>>>(w_out, WOUTT, DD, DD);
  transpose_kernel<<<dim3(8, 16, 1), 256, 0, stream>>>(wq, WQT, DD, 256);
  transpose_kernel<<<dim3(8, 16, 1), 256, 0, stream>>>(wk, WKT, DD, 256);
  transpose_kernel<<<dim3(16, 16, 1), 256, 0, stream>>>(wv_w, WVWT, DD, DD);

  // 1) QKV projection: (10240 x 512) @ (512 x 1536), pad rows < 240 are zero
  GemmP p{};
  p.A = bag; p.sAb = 0; p.lda = DD; p.arow0 = -PAD;
  p.B = WQKVT; p.sBb = 0; p.ldb = DD;
  p.alpha = 1.f; p.M = NPAD; p.N = 3 * DD; p.Kd = DD; p.mode = 1;
  p.Qo = Q; p.Ko = Kb; p.Vo = Vb;
  gemm_bf16_kernel<<<dim3(80, 24, 1), 256, 0, stream>>>(p);

  // V^T per head (for attn3@V GEMM)
  transpose_kernel<<<dim3(2, 320, 8), 256, 0, stream>>>(Vb, VbT, NPAD, DH);

  // 2) landmarks (means over 40 consecutive rows)
  landmark_kernel<<<HEADS * LM * DH / 256, 256, 0, stream>>>(Q, Kb, QL, KL);

  // 3) attn2 = softmax(q_l @ k_l^T)
  attn2_kernel<<<dim3(LM / 16, HEADS), 256, 0, stream>>>(QL, KL, X2);

  // 4) pinv init: z = x^T / (max rowsum * max colsum)
  pinvscale_kernel<<<HEADS * LM, 256, 0, stream>>>(X2, MB);
  zinit_kernel<<<HEADS * LM * LM / 256, 256, 0, stream>>>(X2, MB, Z0);

  // 5) 6 Newton-Schulz iterations, elementwise cI-X folded into matmul epilogues
  float* zc = Z0; float* zn = Z1;
  for (int it = 0; it < 6; ++it) {
    mm256_kernel<<<dim3(2, 16, 8), 256, 0, stream>>>(X2, zc, nullptr, A1, 1.f, 0.f);   // A1 = x@z
    mm256_kernel<<<dim3(2, 16, 8), 256, 0, stream>>>(A1, A1, A1, T1, -1.f, 7.f);       // xz@(7I-xz)
    mm256_kernel<<<dim3(2, 16, 8), 256, 0, stream>>>(A1, T1, A1, T2, -1.f, 15.f);      // xz@(15I-..)
    mm256_kernel<<<dim3(2, 16, 8), 256, 0, stream>>>(zc, T2, zc, zn, -0.25f, 3.25f);   // 0.25*z@(13I-..)
    float* tmp = zc; zc = zn; zn = tmp;
  }

  // 6) attn3 raw logits = q_l @ K^T (batched per head; Kb is already N x K)
  GemmP p3{};
  p3.A = QL; p3.sAb = (long)LM * DH; p3.lda = DH; p3.arow0 = 0;
  p3.B = Kb; p3.sBb = (long)NPAD * DH; p3.ldb = DH;
  p3.C = AT3; p3.sCb = (long)LM * NPAD; p3.ldc = NPAD;
  p3.alpha = 1.f; p3.M = LM; p3.N = NPAD; p3.Kd = DH; p3.mode = 0;
  gemm_bf16_kernel<<<dim3(2, 160, 8), 256, 0, stream>>>(p3);
  rowsm_kernel<<<HEADS * LM, 256, 0, stream>>>(AT3, NPAD);

  // 7) t3 = attn3 @ V  -> stored transposed (64 x 256 per head)
  GemmP p4{};
  p4.A = AT3; p4.sAb = (long)LM * NPAD; p4.lda = NPAD; p4.arow0 = 0;
  p4.B = VbT; p4.sBb = (long)DH * NPAD; p4.ldb = NPAD;
  p4.C = t3T; p4.sCb = (long)DH * LM; p4.ldc = LM; p4.ct = 1;
  p4.alpha = 1.f; p4.M = LM; p4.N = DH; p4.Kd = NPAD; p4.mode = 0;
  gemm_bf16_kernel<<<dim3(2, 1, 8), 256, 0, stream>>>(p4);

  // 8) t4 = pinv(attn2) @ t3 -> stored transposed (64 x 256 per head)
  GemmP p5{};
  p5.A = zc; p5.sAb = (long)LM * LM; p5.lda = LM; p5.arow0 = 0;
  p5.B = t3T; p5.sBb = (long)DH * LM; p5.ldb = LM;
  p5.C = t4T; p5.sCb = (long)DH * LM; p5.ldc = LM; p5.ct = 1;
  p5.alpha = 1.f; p5.M = LM; p5.N = DH; p5.Kd = LM; p5.mode = 0;
  gemm_bf16_kernel<<<dim3(2, 1, 8), 256, 0, stream>>>(p5);

  // 9) fused attn1 softmax + @t4, scatter into (n,512) concat layout (last 10000 rows only)
  attn1_fused_kernel<<<dim3(NN / 16, HEADS), 256, 0, stream>>>(Q, KL, t4T, OC);

  // 10) depthwise residual conv over sequence, += into concat
  convres_kernel<<<NN * DD / 256, 256, 0, stream>>>(Vb, convw, OC);

  // 11) enc = concat @ w_out + b_out + bag
  GemmP p6{};
  p6.A = OC; p6.sAb = 0; p6.lda = DD; p6.arow0 = 0;
  p6.B = WOUTT; p6.sBb = 0; p6.ldb = DD;
  p6.C = ENC; p6.sCb = 0; p6.ldc = DD;
  p6.bias = b_out; p6.E = bag; p6.sEb = 0; p6.ldE = DD; p6.beta = 1.f;
  p6.alpha = 1.f; p6.M = NN; p6.N = DD; p6.Kd = DD; p6.mode = 0;
  gemm_bf16_kernel<<<dim3(79, 8, 1), 256, 0, stream>>>(p6);

  // 12) q2 = enc@wq, k2 = enc@wk
  GemmP p7{};
  p7.A = ENC; p7.sAb = 0; p7.lda = DD; p7.arow0 = 0;
  p7.B = WQT; p7.sBb = 0; p7.ldb = DD;
  p7.C = Q2; p7.sCb = 0; p7.ldc = 256;
  p7.alpha = 1.f; p7.M = NN; p7.N = 256; p7.Kd = DD; p7.mode = 0;
  gemm_bf16_kernel<<<dim3(79, 4, 1), 256, 0, stream>>>(p7);
  p7.B = WKT; p7.C = K2;
  gemm_bf16_kernel<<<dim3(79, 4, 1), 256, 0, stream>>>(p7);

  // 13) sparse edges: A_raw[r] += val * (q2[r].k2[c]) / 16   (A_raw lives in d_out[1..NN])
  edge_kernel<<<((long)E * 32 + 255) / 256, 256, 0, stream>>>(arow, acol, aval, Q2, K2, out + 1, E);

  // 14) alpha = softmax(A_raw)
  alpha_kernel<<<1, 1024, 0, stream>>>(out + 1, ALP);

  // 15) value = bag @ wv_w + wv_b
  GemmP p8{};
  p8.A = bag; p8.sAb = 0; p8.lda = DD; p8.arow0 = 0;
  p8.B = WVWT; p8.sBb = 0; p8.ldb = DD;
  p8.C = VAL; p8.sCb = 0; p8.ldc = DD;
  p8.bias = wv_b;
  p8.alpha = 1.f; p8.M = NN; p8.N = DD; p8.Kd = DD; p8.mode = 0;
  gemm_bf16_kernel<<<dim3(79, 8, 1), 256, 0, stream>>>(p8);

  // 16) xo; pooled = colsum(xo) (k_alpha==1); out = pooled@fc + b
  xo_kernel<<<NN * DD / 256, 256, 0, stream>>>(VAL, ENC, ALP, XO);
  colsum_kernel<<<DD, 256, 0, stream>>>(XO, POOL);
  final_kernel<<<1, 512, 0, stream>>>(POOL, fc_k, fc_b, out);
}
